// CrossAttention_19181323944375
// MI455X (gfx1250) — matrix operand order repaired and checked by reading, not compiled, not run on hardware
//
#include <hip/hip_runtime.h>

// CDNA5 / gfx1250 wave32 WMMA implementation of windowed cross-attention with
// relative position encodings. All GEMM-shaped work goes through
// v_wmma_f32_16x16x32_f16; the i/j-dependent rpe einsums are restructured as
// batched GEMMs over the 2048-window dimension so they also hit the WMMA path.

#define NWIN  2048
#define NHD   12
#define LW    49
#define HDC   32
#define DIMC  384
#define NTOK  100352           // 32*56*56
#define SCALE_F 0.17677669529663687f

typedef _Float16 h16;
typedef __attribute__((ext_vector_type(16))) _Float16 v16h;
typedef __attribute__((ext_vector_type(8)))  _Float16 v8h;
typedef __attribute__((ext_vector_type(4)))  _Float16 v4h;
typedef __attribute__((ext_vector_type(8)))  float    v8f;

#define WMMA_F16(a, b, c) \
  __builtin_amdgcn_wmma_f32_16x16x32_f16(false, (a), false, (b), (short)0, (c), false, false)

// ---------------------------------------------------------------------------
// WMMA fragment loaders (M/N and C/D per CDNA5 ISA §7.12.2; K order is this code's own):
//  A (16x32 f16): lane L holds row M=L&15; K halves selected by L[4];
//                 per lane: contiguous K run of 16, K = (L&16)?16:0 .. +15
//  B (32x16 f16): lane L holds col N=L&15; contiguous K run of 16,
//                 K = (L&16)?16:0 .. +15
//  C/D (16x16 f32, v8f): element r -> M = r + ((L&16)?8:0), N = L&15
// Each fragment = two 16-byte loads per lane.
// ---------------------------------------------------------------------------
__device__ __forceinline__ v16h loadA(const h16* base, int pitch, int k0, int mmax) {
  int ln = threadIdx.x & 31;
  int m = ln & 15; if (m > mmax) m = mmax;
  const h16* p = base + m * pitch + k0 + ((ln & 16) ? 16 : 0);
  v8h lo = *(const v8h*)(p);
  v8h hi = *(const v8h*)(p + 8);
  v16h r;
#pragma unroll
  for (int e = 0; e < 8; ++e) { r[e] = lo[e]; r[e + 8] = hi[e]; }
  return r;
}

__device__ __forceinline__ v16h loadB(const h16* base, int pitch, int k0, int nmax) {
  int ln = threadIdx.x & 31;
  int n = ln & 15; if (n > nmax) n = nmax;
  const h16* p = base + n * pitch + k0 + ((ln & 16) ? 16 : 0);
  v8h lo = *(const v8h*)(p);
  v8h hi = *(const v8h*)(p + 8);
  v16h r;
#pragma unroll
  for (int e = 0; e < 8; ++e) { r[e] = lo[e]; r[e + 8] = hi[e]; }
  return r;
}

__device__ __forceinline__ int tok_of(int win, int l) {
  int b = win >> 6, wi = (win >> 3) & 7, wj = win & 7;
  int hh = wi * 7 + l / 7, ww = wj * 7 + l % 7;
  return (b * 56 + hh) * 56 + ww;
}

__device__ __forceinline__ int rpi_of(int i, int j) {
  return (i / 7 - j / 7 + 6) * 13 + (i % 7 - j % 7 + 6);
}

// ---------------------------------------------------------------------------
// Prep 1: transpose weights to [N][K] f16 so WMMA B-fragments read global directly
// ---------------------------------------------------------------------------
__global__ __launch_bounds__(256) void prep_weights(const float* qw, const float* kvw,
                                                    const float* pw,
                                                    h16* wqT, h16* wkvT, h16* wpT) {
  const int n0 = DIMC * DIMC, n1 = DIMC * 2 * DIMC;
  int total = n0 + n1 + n0;
  for (int idx = blockIdx.x * blockDim.x + threadIdx.x; idx < total;
       idx += gridDim.x * blockDim.x) {
    if (idx < n0) {
      int n = idx / DIMC, k = idx % DIMC;
      wqT[idx] = (h16)qw[k * DIMC + n];
    } else if (idx < n0 + n1) {
      int t = idx - n0;
      int n = t / DIMC, k = t % DIMC;
      wkvT[t] = (h16)kvw[k * (2 * DIMC) + n];
    } else {
      int t = idx - n0 - n1;
      int n = t / DIMC, k = t % DIMC;
      wpT[t] = (h16)pw[k * DIMC + n];
    }
  }
}

// ---------------------------------------------------------------------------
// Prep 2: gather rpe_table by the (recomputed) relative-position index into
//   krpe [h][i][j][c]           (B for qr GEMMs, N=j, K=c)
//   qrpe [h][j][i][c] * SCALE   (B for kr GEMMs, N=i, K=c)
//   vrpe [h][i][c][j64]         (B for out2 GEMMs, N=c, K=j; j padded 64 w/ 0)
// ---------------------------------------------------------------------------
__global__ __launch_bounds__(256) void prep_rpe(const float* rpe,
                                                h16* krpe, h16* qrpe, h16* vrpe) {
  const int nk = NHD * LW * LW * HDC;        // 921984
  const int nv = NHD * LW * HDC * 64;        // 1204224
  int total = 2 * nk + nv;
  for (int idx = blockIdx.x * blockDim.x + threadIdx.x; idx < total;
       idx += gridDim.x * blockDim.x) {
    if (idx < nk) {
      int c = idx & 31; int rest = idx >> 5;
      int j = rest % LW; int i = (rest / LW) % LW; int h = rest / (LW * LW);
      krpe[idx] = (h16)rpe[rpi_of(i, j) * (3 * DIMC) + h * 96 + 32 + c];
    } else if (idx < 2 * nk) {
      int t = idx - nk;
      int c = t & 31; int rest = t >> 5;
      int i = rest % LW; int j = (rest / LW) % LW; int h = rest / (LW * LW);
      qrpe[t] = (h16)(rpe[rpi_of(i, j) * (3 * DIMC) + h * 96 + c] * SCALE_F);
    } else {
      int t = idx - 2 * nk;
      int j = t & 63; int rest = t >> 6;
      int c = rest & 31; int hi = rest >> 5;
      int i = hi % LW; int h = hi / LW;
      vrpe[t] = (j < LW) ? (h16)rpe[rpi_of(i, j) * (3 * DIMC) + h * 96 + 64 + c]
                         : (h16)0.0f;
    }
  }
}

// ---------------------------------------------------------------------------
// QKV projection GEMM. Block = 4 waves; shared 16x32 A tile (f32->f16 in LDS),
// each wave owns a 16-wide N strip. A-tile staging is closed-form: each of the
// 128 threads owns exactly one float4 load -> packed f16 -> one 8B LDS store.
// Output is written directly in the window-partitioned layout [win][h][l][c]
// (q pre-scaled by SCALE). grid.y < 6: q job; else kv job.
// ---------------------------------------------------------------------------
__global__ __launch_bounds__(128) void qkv_kernel(const float* x, const float* ctx,
                                                  const float* qb, const float* kvb,
                                                  const h16* wqT, const h16* wkvT,
                                                  h16* q16, h16* k16, h16* v16) {
  __shared__ __attribute__((aligned(16))) h16 At[16 * 32];
  int tid = threadIdx.x, wave = tid >> 5, ln = tid & 31;
  int m0 = blockIdx.x * 16;
  bool isQ = blockIdx.y < 6;
  int n0 = (isQ ? blockIdx.y : blockIdx.y - 6) * 64;
  const float* A = isQ ? x : ctx;
  const h16* WT = isQ ? wqT : wkvT;
  // this thread's fixed slot in the 16x32 staging tile (4 f16 elements)
  const int sm = tid >> 3;            // row 0..15
  const int sk = (tid & 7) * 4;       // col 0,4,...,28
  const float* Asrc = A + (size_t)(m0 + sm) * DIMC + sk;
  v8f acc = {0.f, 0.f, 0.f, 0.f, 0.f, 0.f, 0.f, 0.f};
  for (int k0 = 0; k0 < DIMC; k0 += 32) {
    __syncthreads();
    {
      const float4 f = *(const float4*)(Asrc + k0);
      v4h o;
      o[0] = (h16)f.x; o[1] = (h16)f.y; o[2] = (h16)f.z; o[3] = (h16)f.w;
      *(v4h*)&At[sm * 32 + sk] = o;
    }
    if (k0 + 32 < DIMC)
      __builtin_prefetch(Asrc + k0 + 32, 0, 1);
    __syncthreads();
    v16h a = loadA(At, 32, 0, 15);
    v16h b = loadB(WT + (size_t)(n0 + wave * 16) * DIMC, DIMC, k0, 15);
    acc = WMMA_F16(a, b, acc);
  }
  int nn = n0 + wave * 16 + (ln & 15);
  int mb = m0 + ((ln & 16) ? 8 : 0);
#pragma unroll
  for (int r = 0; r < 8; ++r) {
    int t = mb + r;
    int bb = t / 3136, rr = t % 3136, hh = rr / 56, ww = rr % 56;
    int win = bb * 64 + (hh / 7) * 8 + (ww / 7);
    int l = (hh % 7) * 7 + (ww % 7);
    if (isQ) {
      int h = nn >> 5, c = nn & 31;
      q16[(((size_t)win * NHD + h) * LW + l) * HDC + c] =
          (h16)((acc[r] + qb[nn]) * SCALE_F);
    } else {
      float v = acc[r] + kvb[nn];
      int c = nn & 31;
      if (nn < DIMC) {
        int h = nn >> 5;
        k16[(((size_t)win * NHD + h) * LW + l) * HDC + c] = (h16)v;
      } else {
        int h = (nn - DIMC) >> 5;
        v16[(((size_t)win * NHD + h) * LW + l) * HDC + c] = (h16)v;
      }
    }
  }
}

// ---------------------------------------------------------------------------
// qr: for fixed (h,i), rb[b,h,i,j] = Q[b,:] @ k_rpe[i,j,:]^T  -- GEMM M=2048
// windows, K=32, N=49. A read straight from global q16 (row stride = window
// stride). grid = (2048/64, 12*49); 4 waves split M.
// ---------------------------------------------------------------------------
__global__ __launch_bounds__(128) void qr_kernel(const h16* q16, const h16* krpe,
                                                 h16* rb) {
  int hi = blockIdx.y, h = hi / LW, i = hi % LW;
  int wave = threadIdx.x >> 5, ln = threadIdx.x & 31;
  int b0 = blockIdx.x * 64 + wave * 16;
  const h16* Ab = q16 + (((size_t)b0 * NHD + h) * LW + i) * HDC;
  const int apitch = NHD * LW * HDC;
  v16h a = loadA(Ab, apitch, 0, 15);
  const h16* Bb = krpe + (size_t)hi * LW * HDC;
#pragma unroll
  for (int jt = 0; jt < 4; ++jt) {
    int j0 = jt * 16;
    int nmax = LW - 1 - j0; if (nmax > 15) nmax = 15;
    v16h b = loadB(Bb + j0 * HDC, HDC, 0, nmax);
    v8f acc = {0.f, 0.f, 0.f, 0.f, 0.f, 0.f, 0.f, 0.f};
    acc = WMMA_F16(a, b, acc);
    int j = j0 + (ln & 15);
    if (j < LW) {
      int mb = b0 + ((ln & 16) ? 8 : 0);
#pragma unroll
      for (int r = 0; r < 8; ++r)
        rb[(((size_t)(mb + r) * NHD + h) * LW + i) * LW + j] = (h16)acc[r];
    }
  }
}

// ---------------------------------------------------------------------------
// kr: for fixed (h,j), rb[b,h,i,j] += K[b,:] @ q_rpe_scaled[i,j,:]^T (N=i).
// Race-free element-wise f16 accumulate (unique owner per element; stream
// ordering after qr_kernel guarantees the '=' happened first).
// ---------------------------------------------------------------------------
__global__ __launch_bounds__(128) void kr_kernel(const h16* k16, const h16* qrpe,
                                                 h16* rb) {
  int hj = blockIdx.y, h = hj / LW, j = hj % LW;
  int wave = threadIdx.x >> 5, ln = threadIdx.x & 31;
  int b0 = blockIdx.x * 64 + wave * 16;
  const h16* Ab = k16 + (((size_t)b0 * NHD + h) * LW + j) * HDC;
  const int apitch = NHD * LW * HDC;
  v16h a = loadA(Ab, apitch, 0, 15);
  const h16* Bb = qrpe + (size_t)hj * LW * HDC;
#pragma unroll
  for (int it = 0; it < 4; ++it) {
    int i0 = it * 16;
    int nmax = LW - 1 - i0; if (nmax > 15) nmax = 15;
    v16h b = loadB(Bb + i0 * HDC, HDC, 0, nmax);
    v8f acc = {0.f, 0.f, 0.f, 0.f, 0.f, 0.f, 0.f, 0.f};
    acc = WMMA_F16(a, b, acc);
    int i = i0 + (ln & 15);
    if (i < LW) {
      int mb = b0 + ((ln & 16) ? 8 : 0);
#pragma unroll
      for (int r = 0; r < 8; ++r) {
        size_t ad = (((size_t)(mb + r) * NHD + h) * LW + i) * LW + j;
        rb[ad] = (h16)((float)rb[ad] + acc[r]);
      }
    }
  }
}

// ---------------------------------------------------------------------------
// Per-(window,head) attention: S = qk (WMMA, C-init from rb, -1e30 in j-pad),
// row softmax in LDS, O1 = P@V (WMMA). P also spilled to global (padded to 64
// cols) for the v_rpe pass. 4 waves each own a 16-row i strip. All LDS staging
// is closed-form / fully vectorized (no dynamic grid-stride loops).
// ---------------------------------------------------------------------------
__global__ __launch_bounds__(128) void attn_kernel(const h16* q16, const h16* k16,
                                                   const h16* v16, const h16* rb,
                                                   h16* attn_g, h16* outTok) {
  __shared__ __attribute__((aligned(16))) h16 ks[64 * 32];   // [j64][c]  (B for qk)
  __shared__ __attribute__((aligned(16))) h16 vT[32 * 64];   // [c][j64]  (B for P@V)
  __shared__ __attribute__((aligned(16))) float S[64 * 64];  // logits
  __shared__ __attribute__((aligned(16))) h16 P[64 * 64];    // softmax probs f16
  int win = blockIdx.x / NHD, h = blockIdx.x % NHD;
  int tid = threadIdx.x, wave = tid >> 5, ln = tid & 31;
  size_t base = ((size_t)win * NHD + h) * LW * HDC;

  {  // ks: each thread owns half a row (16 f16 = 32B): 2x b128 copy or zero-fill
    int j = tid >> 1, half = tid & 1;
    h16* dst = ks + j * HDC + half * 16;
    if (j < LW) {
      const h16* src = k16 + base + j * HDC + half * 16;
      *(v8h*)dst = *(const v8h*)src;
      *(v8h*)(dst + 8) = *(const v8h*)(src + 8);
    } else {
      v8h z = {(h16)0, (h16)0, (h16)0, (h16)0, (h16)0, (h16)0, (h16)0, (h16)0};
      *(v8h*)dst = z;
      *(v8h*)(dst + 8) = z;
    }
  }
  {  // vT (transpose): thread owns (c, 16-wide j strip); fully unrolled
    int c = tid >> 2, j0 = (tid & 3) * 16;
    h16* dst = vT + c * 64 + j0;
#pragma unroll
    for (int u = 0; u < 16; ++u) {
      int j = j0 + u;
      dst[u] = (j < LW) ? v16[base + j * HDC + c] : (h16)0.0f;
    }
  }
  __syncthreads();

  int i0 = wave * 16;
  int mmax = LW - 1 - i0; if (mmax > 15) mmax = 15;
  v16h a = loadA(q16 + base + i0 * HDC, HDC, 0, mmax);
  const h16* rbb = rb + ((size_t)win * NHD + h) * LW * LW;
#pragma unroll
  for (int jt = 0; jt < 4; ++jt) {
    int j0 = jt * 16;
    v16h b = loadB(ks + j0 * HDC, HDC, 0, 15);      // zero-padded rows j>=49
    int jj = j0 + (ln & 15);
    int mb = (ln & 16) ? 8 : 0;
    v8f c;
#pragma unroll
    for (int r = 0; r < 8; ++r) {
      int ii = i0 + mb + r;
      c[r] = (jj < LW && ii < LW) ? (float)rbb[ii * LW + jj] : -1e30f;
    }
    c = WMMA_F16(a, b, c);
#pragma unroll
    for (int r = 0; r < 8; ++r) S[(i0 + mb + r) * 64 + jj] = c[r];
  }
  __syncthreads();

  if (tid < LW) {
    int i = tid;
    float mx = -1e30f;
#pragma unroll
    for (int j = 0; j < LW; ++j) mx = fmaxf(mx, S[i * 64 + j]);
    float sum = 0.f;
#pragma unroll
    for (int j = 0; j < LW; ++j) {
      float e = __expf(S[i * 64 + j] - mx);
      S[i * 64 + j] = e;
      sum += e;
    }
    float inv = 1.0f / sum;
    h16* grow = attn_g + ((size_t)(h * LW + i) * NWIN + win) * 64;
#pragma unroll
    for (int j = 0; j < 64; ++j) {
      h16 p = (j < LW) ? (h16)(S[i * 64 + j] * inv) : (h16)0.0f;
      P[i * 64 + j] = p;
      grow[j] = p;
    }
  } else if (tid < 64) {
    v8h z = {(h16)0, (h16)0, (h16)0, (h16)0, (h16)0, (h16)0, (h16)0, (h16)0};
#pragma unroll
    for (int j = 0; j < 64; j += 8) *(v8h*)(P + tid * 64 + j) = z;  // pad rows
  }
  __syncthreads();

  v8f oa[2] = {{0.f, 0.f, 0.f, 0.f, 0.f, 0.f, 0.f, 0.f},
               {0.f, 0.f, 0.f, 0.f, 0.f, 0.f, 0.f, 0.f}};
#pragma unroll
  for (int k0 = 0; k0 < 64; k0 += 32) {
    v16h af = loadA(P + i0 * 64, 64, k0, 15);
#pragma unroll
    for (int ct = 0; ct < 2; ++ct) {
      v16h bf = loadB(vT + ct * 16 * 64, 64, k0, 15);
      oa[ct] = WMMA_F16(af, bf, oa[ct]);
    }
  }
  int mb = (ln & 16) ? 8 : 0;
#pragma unroll
  for (int ct = 0; ct < 2; ++ct) {
    int c = ct * 16 + (ln & 15);
#pragma unroll
    for (int r = 0; r < 8; ++r) {
      int i = i0 + mb + r;
      if (i < LW) {
        int t = tok_of(win, i);
        outTok[(size_t)t * DIMC + h * HDC + c] = (h16)oa[ct][r];
      }
    }
  }
}

// ---------------------------------------------------------------------------
// out2: for fixed (h,i), O[b,:] += attn[b,h,i,:] @ v_rpe[i,:,h,:]  -- GEMM
// M=2048 windows, K=49 (padded 64, zeros), N=32. Accumulates into outTok.
// ---------------------------------------------------------------------------
__global__ __launch_bounds__(128) void out2_kernel(const h16* attn_g, const h16* vrpe,
                                                   h16* outTok) {
  int hi = blockIdx.y, h = hi / LW, i = hi % LW;
  int wave = threadIdx.x >> 5, ln = threadIdx.x & 31;
  int b0 = blockIdx.x * 64 + wave * 16;
  const h16* Ab = attn_g + ((size_t)hi * NWIN + b0) * 64;
  const h16* Bb = vrpe + (size_t)hi * HDC * 64;
  v8f acc[2] = {{0.f, 0.f, 0.f, 0.f, 0.f, 0.f, 0.f, 0.f},
                {0.f, 0.f, 0.f, 0.f, 0.f, 0.f, 0.f, 0.f}};
#pragma unroll
  for (int k0 = 0; k0 < 64; k0 += 32) {
    v16h a = loadA(Ab, 64, k0, 15);
#pragma unroll
    for (int ct = 0; ct < 2; ++ct) {
      v16h b = loadB(Bb + ct * 16 * 64, 64, k0, 15);
      acc[ct] = WMMA_F16(a, b, acc[ct]);
    }
  }
  int mb = b0 + ((ln & 16) ? 8 : 0);
#pragma unroll
  for (int ct = 0; ct < 2; ++ct) {
    int c = ct * 16 + (ln & 15);
#pragma unroll
    for (int r = 0; r < 8; ++r) {
      int t = tok_of(mb + r, i);
      size_t ad = (size_t)t * DIMC + h * HDC + c;
      outTok[ad] = (h16)((float)outTok[ad] + acc[ct][r]);
    }
  }
}

// ---------------------------------------------------------------------------
// Output projection: [100352 x 384] @ [384 x 384] + bias -> f32 d_out.
// A (token-major f16) and B (pre-transposed f16) both fragment-loaded from
// global; 4 waves split a 64-wide N strip.
// ---------------------------------------------------------------------------
__global__ __launch_bounds__(128) void proj_kernel(const h16* outTok, const h16* wpT,
                                                   const float* pb, float* out) {
  int wave = threadIdx.x >> 5, ln = threadIdx.x & 31;
  int m0 = blockIdx.x * 16;
  int n0 = blockIdx.y * 64 + wave * 16;
  const h16* Ab = outTok + (size_t)m0 * DIMC;
  const h16* Bb = wpT + (size_t)n0 * DIMC;
  v8f acc = {0.f, 0.f, 0.f, 0.f, 0.f, 0.f, 0.f, 0.f};
#pragma unroll
  for (int k0 = 0; k0 < DIMC; k0 += 32) {
    v16h a = loadA(Ab, DIMC, k0, 15);
    v16h b = loadB(Bb, DIMC, k0, 15);
    acc = WMMA_F16(a, b, acc);
  }
  int n = n0 + (ln & 15);
  int mb = m0 + ((ln & 16) ? 8 : 0);
  float bias = pb[n];
#pragma unroll
  for (int r = 0; r < 8; ++r)
    out[(size_t)(mb + r) * DIMC + n] = acc[r] + bias;
}

// ---------------------------------------------------------------------------
extern "C" void kernel_launch(void* const* d_in, const int* in_sizes, int n_in,
                              void* d_out, int out_size, void* d_ws, size_t ws_size,
                              hipStream_t stream) {
  const float* x   = (const float*)d_in[0];
  const float* ctx = (const float*)d_in[1];
  const float* rpe = (const float*)d_in[2];
  const float* qw  = (const float*)d_in[3];
  const float* qb  = (const float*)d_in[4];
  const float* kvw = (const float*)d_in[5];
  const float* kvb = (const float*)d_in[6];
  const float* pw  = (const float*)d_in[7];
  const float* pb  = (const float*)d_in[8];
  float* out = (float*)d_out;

  char* ws = (char*)d_ws;
  size_t off = 0;
  auto carve = [&](size_t bytes) -> char* {
    char* p = ws + off;
    off += (bytes + 255) & ~(size_t)255;
    return p;
  };
  const size_t QKV_E = (size_t)NWIN * NHD * LW * HDC;     // 38,535,168
  h16* q16    = (h16*)carve(QKV_E * 2);
  h16* k16    = (h16*)carve(QKV_E * 2);
  h16* v16    = (h16*)carve(QKV_E * 2);
  h16* wqT    = (h16*)carve((size_t)DIMC * DIMC * 2);
  h16* wkvT   = (h16*)carve((size_t)DIMC * 2 * DIMC * 2);
  h16* wpT    = (h16*)carve((size_t)DIMC * DIMC * 2);
  h16* krpe   = (h16*)carve((size_t)NHD * LW * LW * HDC * 2);
  h16* qrpe   = (h16*)carve((size_t)NHD * LW * LW * HDC * 2);
  h16* vrpe   = (h16*)carve((size_t)NHD * LW * HDC * 64 * 2);
  h16* rb     = (h16*)carve((size_t)NWIN * NHD * LW * LW * 2);   // 118 MB
  h16* attn_g = (h16*)carve((size_t)NHD * LW * NWIN * 64 * 2);   // 154 MB
  h16* outTok = (h16*)carve((size_t)NTOK * DIMC * 2);            // 77 MB
  (void)ws_size; (void)in_sizes; (void)n_in; (void)out_size;

  prep_weights<<<1024, 256, 0, stream>>>(qw, kvw, pw, wqT, wkvT, wpT);
  prep_rpe<<<2048, 256, 0, stream>>>(rpe, krpe, qrpe, vrpe);
  qkv_kernel<<<dim3(NTOK / 16, 18), 128, 0, stream>>>(x, ctx, qb, kvb, wqT, wkvT,
                                                      q16, k16, v16);
  qr_kernel<<<dim3(NWIN / 64, NHD * LW), 128, 0, stream>>>(q16, krpe, rb);
  kr_kernel<<<dim3(NWIN / 64, NHD * LW), 128, 0, stream>>>(k16, qrpe, rb);
  attn_kernel<<<NWIN * NHD, 128, 0, stream>>>(q16, k16, v16, rb, attn_g, outTok);
  out2_kernel<<<dim3(NWIN / 64, NHD * LW), 128, 0, stream>>>(attn_g, vrpe, outTok);
  proj_kernel<<<dim3(NTOK / 16, 6), 128, 0, stream>>>(outTok, wpT, pb, out);
}
